// Swin3DBlock_78735340470527
// MI455X (gfx1250) — matrix operand order repaired and checked by reading, not compiled, not run on hardware
//
#include <hip/hip_runtime.h>
#include <hip/hip_bf16.h>
#include <math.h>

typedef __attribute__((ext_vector_type(16))) _Float16 v16h;
typedef __attribute__((ext_vector_type(8)))  float    v8f;
typedef unsigned int uint4v __attribute__((ext_vector_type(4)));
typedef int          int8v  __attribute__((ext_vector_type(8)));
typedef int          int4v  __attribute__((ext_vector_type(4)));

#define WAVES 8

#if defined(__has_builtin)
#if __has_builtin(__builtin_amdgcn_tensor_load_to_lds) && __has_builtin(__builtin_amdgcn_s_wait_tensorcnt)
#define HAVE_TDM 1
#endif
#endif
#ifndef HAVE_TDM
#define HAVE_TDM 0
#endif

// ---------------- WMMA fragment loaders (CDNA5 wave32 layouts) ----------------
__device__ __forceinline__ v16h ldA(const _Float16* __restrict__ base, int lda, int lane) {
  int row = lane & 15;
  int kh  = (lane >> 4) << 3;  // 0 or 8
  const _Float16* p = base + row * lda + kh;
  v16h a;
#pragma unroll
  for (int e = 0; e < 8; ++e) a[e] = p[e];
#pragma unroll
  for (int e = 0; e < 8; ++e) a[8 + e] = p[16 + e];
  return a;
}
// B source stored column-major as Bt[n][k] (K contiguous), ldb in halves.
__device__ __forceinline__ v16h ldB(const _Float16* __restrict__ base, int ldb, int lane) {
  int col = lane & 15;
  int kb  = (lane >> 4) << 3;  // 0 or 8
  const _Float16* p = base + col * ldb + kb;
  v16h b;
#pragma unroll
  for (int e = 0; e < 8; ++e) b[e] = p[e];
#pragma unroll
  for (int e = 0; e < 8; ++e) b[8 + e] = p[16 + e];
  return b;
}

#define WMMA_F16(A_, B_, C_) \
  __builtin_amdgcn_wmma_f32_16x16x32_f16(false, (A_), false, (B_), (short)0, (C_), false, false)

// ---------------- token geometry ----------------
__device__ __forceinline__ int tok_region(int wdi, int whi, int wwi, int n) {
  int od = n / 49; int r = n - od * 49; int oh = r / 7; int ow = r - oh * 7;
  int gd = wdi * 2 + od, gh = whi * 7 + oh, gw = wwi * 7 + ow;
  int fd = (gd < 14) ? 0 : ((gd < 15) ? 1 : 2);
  int fh = (gh < 49) ? 0 : ((gh < 53) ? 1 : 2);
  int fw = (gw < 49) ? 0 : ((gw < 53) ? 1 : 2);
  return fd * 9 + fh * 3 + fw;
}
__device__ __forceinline__ int tok_base(int b, int wdi, int whi, int wwi, int n) {
  int od = n / 49; int r = n - od * 49; int oh = r / 7; int ow = r - oh * 7;
  int gd = wdi * 2 + od, gh = whi * 7 + oh, gw = wwi * 7 + ow;
  int pd = gd + 1; if (pd >= 16) pd -= 16;
  int ph = gh + 3; if (ph >= 56) ph -= 56;
  int pw = gw + 3; if (pw >= 56) pw -= 56;
  return b * 4816896 + pd * 3136 + ph * 56 + pw;   // + c*50176 per channel
}

// ---------------- weight prep: fp32 -> f16, transposed to [out][in] ----------------
__global__ void swin_prep_weights(const float* __restrict__ qkv_w,   // (96,288)
                                  const float* __restrict__ proj_w,  // (96,96)
                                  const float* __restrict__ fc1_w,   // (96,384)
                                  const float* __restrict__ fc2_w,   // (384,96)
                                  _Float16* __restrict__ wt) {
  int i = blockIdx.x * 256 + threadIdx.x;
  if (i < 288 * 96) { int n = i / 96, k = i % 96; wt[i] = (_Float16)qkv_w[k * 288 + n]; return; }
  i -= 288 * 96;
  if (i < 96 * 96)  { int n = i / 96, k = i % 96; wt[27648 + i] = (_Float16)proj_w[k * 96 + n]; return; }
  i -= 96 * 96;
  if (i < 384 * 96) { int n = i / 96, k = i % 96; wt[36864 + i] = (_Float16)fc1_w[k * 384 + n]; return; }
  i -= 384 * 96;
  if (i < 96 * 384) { int n = i / 384, k = i % 384; wt[73728 + i] = (_Float16)fc2_w[k * 96 + n]; return; }
}

// ---------------- fused Swin3D block: 1 workgroup per window ----------------
__global__ __launch_bounds__(256)
void swin3d_block(const float* __restrict__ x,
                  const float* __restrict__ g1, const float* __restrict__ b1,
                  const float* __restrict__ g2, const float* __restrict__ b2,
                  const float* __restrict__ qkv_b,
                  const float* __restrict__ proj_b,
                  const float* __restrict__ rpb_table,   // (507,4)
                  const float* __restrict__ fc1_b,
                  const float* __restrict__ fc2_b,
                  const _Float16* __restrict__ wqkv,     // [288][96]
                  const _Float16* __restrict__ wproj,    // [96][96]
                  const _Float16* __restrict__ wfc1,     // [384][96]
                  const _Float16* __restrict__ wfc2,     // [96][384]
                  float* __restrict__ out) {
  extern __shared__ char smem[];
  _Float16* ALN = (_Float16*)(smem + 0);        // 112 x 96  (A for qkv)
  _Float16* Q   = (_Float16*)(smem + 21504);    // 4 x 112 x 32
  _Float16* Kb  = (_Float16*)(smem + 50176);    // 4 x 112 x 32
  _Float16* VT  = (_Float16*)(smem + 78848);    // 4 x 32 x 128  (v transposed)
  float*    S   = (float*)   (smem + 111616);   // 112 x 112
  _Float16* P   = (_Float16*)(smem + 161792);   // 112 x 128
  float*    RPB = (float*)   (smem + 190464);   // 507 x 4 = 8112 B
  int*      RPOFF = (int*)   (smem + 198592);   // 112
  int*      REGT  = (int*)   (smem + 199040);   // 112
  // --- MLP-phase aliases (attention buffers dead by then) ---
  _Float16* AC  = (_Float16*)(smem + 0);        // attn concat 112 x 96
  float*    Y   = (float*)   (smem + 21504);    // residual-1 112 x 96 f32
  _Float16* H2  = (_Float16*)(smem + 64512);    // LN2 out 112 x 96
  _Float16* F1  = (_Float16*)(smem + 86016);    // fc1+gelu 112 x 384

  const int tid = threadIdx.x;
  const int lane = tid & 31;
  const int wave = tid >> 5;
  const int b    = blockIdx.x >> 9;
  const int widx = blockIdx.x & 511;
  const int wdi = widx >> 6, whi = (widx >> 3) & 7, wwi = widx & 7;

#if HAVE_TDM
  // ---- TDM: async DMA of rel-pos-bias table (507x4 f32, 1D tile) into LDS ----
  if (wave == 0) {
    unsigned long long ga = (unsigned long long)(const void*)rpb_table;
    uint4v g0;
    g0[0] = 1u;                                   // count=1 (valid user descriptor)
    g0[1] = 190464u;                              // lds_addr (bytes)
    g0[2] = (unsigned int)(ga & 0xffffffffu);     // global_addr[31:0]
    g0[3] = (unsigned int)((ga >> 32) & 0x01ffffffu) | (2u << 30);  // addr[56:32] | type=2
    int8v g1;
    g1[0] = 2 << 16;                              // data_size=4B; no multicast/pad/iterate
    g1[1] = (2028 & 0xffff) << 16;                // tensor_dim0[15:0]
    g1[2] = (2028 >> 16) | (1 << 16);             // tensor_dim0[31:16] | tensor_dim1[15:0]=1
    g1[3] = (2028 & 0xffff) << 16;                // tile_dim0 = 2028 elements
    g1[4] = 0;                                    // tile_dim1=0, tile_dim2=0
    g1[5] = 2028;                                 // tensor_dim0_stride
    g1[6] = 0; g1[7] = 0;
    int4v z4 = {0, 0, 0, 0};
    int8v z8 = {0, 0, 0, 0, 0, 0, 0, 0};
    __builtin_amdgcn_tensor_load_to_lds(g0, g1, z4, z4, z8, 0);
  }
#endif

  // zero-init padded staging (covers K-dim and token padding)
  for (int i = tid; i < 190464 / 4; i += 256) ((float*)smem)[i] = 0.0f;
#if !HAVE_TDM
  for (int i = tid; i < 507 * 4; i += 256) RPB[i] = rpb_table[i];
#endif
  // per-token relpos offset + mask-region tables
  if (tid < 112) {
    int n = tid; int od = n / 49; int r = n - od * 49; int oh = r / 7; int ow = r - oh * 7;
    RPOFF[n] = od * 169 + oh * 13 + ow;
    REGT[n]  = tok_region(wdi, whi, wwi, n);
  }
#if HAVE_TDM
  __builtin_amdgcn_s_wait_tensorcnt(0);
#endif
  __syncthreads();

  // ---- phase 1: LN1 (gather channels at shifted-source coordinate) ----
  if (tid < 98) {
    int base = tok_base(b, wdi, whi, wwi, tid);
    float s1 = 0.f, s2 = 0.f;
    for (int c = 0; c < 96; ++c) { float v = x[base + c * 50176]; s1 += v; s2 += v * v; }
    float mu = s1 * (1.0f / 96.0f);
    float var = s2 * (1.0f / 96.0f) - mu * mu;
    float rstd = rsqrtf(var + 1e-5f);
    for (int c = 0; c < 96; ++c) {
      float v = x[base + c * 50176];
      ALN[tid * 96 + c] = (_Float16)((v - mu) * rstd * g1[c] + b1[c]);
    }
  }
  __syncthreads();

  // ---- phase 2: qkv GEMM (112x96)@(96x288), scatter into q/k/v^T per head ----
  for (int t = wave; t < 7 * 18; t += WAVES) {
    int mt = t / 18, nt = t % 18;
    v16h af[3], bf[3];
#pragma unroll
    for (int ks = 0; ks < 3; ++ks) {
      af[ks] = ldA(ALN + mt * 16 * 96 + ks * 32, 96, lane);
      bf[ks] = ldB(wqkv + (nt * 16) * 96 + ks * 32, 96, lane);
    }
    v8f c = {};
#pragma unroll
    for (int ks = 0; ks < 3; ++ks) c = WMMA_F16(af[ks], bf[ks], c);

    int n0 = mt * 16 + ((lane >> 4) << 3);
    int j  = nt * 16 + (lane & 15);
    float bias = qkv_b[j];
    int s = nt / 6;                                 // wave-uniform: which of q/k/v
    int rem = j - s * 96; int hh = rem / 24; int dd = rem - hh * 24;
    if (s == 0) {
#pragma unroll
      for (int r = 0; r < 8; ++r)
        Q[(hh * 112 + n0 + r) * 32 + dd] = (_Float16)((c[r] + bias) * 0.2041241452f);
    } else if (s == 1) {
#pragma unroll
      for (int r = 0; r < 8; ++r)
        Kb[(hh * 112 + n0 + r) * 32 + dd] = (_Float16)(c[r] + bias);
    } else {
#pragma unroll
      for (int r = 0; r < 8; ++r)
        VT[(hh * 32 + dd) * 128 + n0 + r] = (_Float16)(c[r] + bias);
    }
  }

  // warm the residual-x lines (consumed in phase 4) behind 4 heads of attention
  if (tid < 98) {
    int base = tok_base(b, wdi, whi, wwi, tid);
    for (int c = 0; c < 96; ++c) __builtin_prefetch(&x[base + c * 50176], 0, 1);
  }
  __syncthreads();

  // ---- phase 3: per-head attention ----
  for (int h = 0; h < 4; ++h) {
    const _Float16* Qh = Q  + h * 112 * 32;
    const _Float16* Kh = Kb + h * 112 * 32;
    const _Float16* Vh = VT + h * 32 * 128;
    // S = q @ k^T  (K = 32, dh padded)
    for (int t = wave; t < 7 * 7; t += WAVES) {
      int mt = t / 7, nt = t % 7;
      v16h a  = ldA(Qh + mt * 16 * 32, 32, lane);
      v16h bm = ldB(Kh + nt * 16 * 32, 32, lane);
      v8f c = {};
      c = WMMA_F16(a, bm, c);
      int n0 = mt * 16 + ((lane >> 4) << 3);
      int m  = nt * 16 + (lane & 15);
#pragma unroll
      for (int r = 0; r < 8; ++r) S[(n0 + r) * 112 + m] = c[r];
    }
    __syncthreads();
    // softmax with rel-pos-bias + shift mask, write P (f16, K padded to 128)
    if (tid < 112) {
      int n = tid;
      if (n < 98) {
        int rpibase = RPOFF[n] + 253;   // +1*169 + 6*13 + 6
        int regn = REGT[n];
        float mx = -1e30f;
        for (int m = 0; m < 98; ++m) {
          float w = S[n * 112 + m] + RPB[(rpibase - RPOFF[m]) * 4 + h];
          if (REGT[m] != regn) w -= 100.0f;
          S[n * 112 + m] = w;
          mx = fmaxf(mx, w);
        }
        float sum = 0.f;
        for (int m = 0; m < 98; ++m) { float e = __expf(S[n * 112 + m] - mx); S[n * 112 + m] = e; sum += e; }
        float inv = 1.0f / sum;
        for (int m = 0; m < 98; ++m)  P[n * 128 + m] = (_Float16)(S[n * 112 + m] * inv);
        for (int m = 98; m < 128; ++m) P[n * 128 + m] = (_Float16)0.0f;
      } else {
        for (int m = 0; m < 128; ++m) P[n * 128 + m] = (_Float16)0.0f;
      }
    }
    __syncthreads();
    // out_h = P @ v  (K padded 98->128), concat into AC
    for (int t = wave; t < 7 * 2; t += WAVES) {
      int mt = t >> 1, nt = t & 1;
      v16h af[4], bf[4];
#pragma unroll
      for (int ks = 0; ks < 4; ++ks) {
        af[ks] = ldA(P + mt * 16 * 128 + ks * 32, 128, lane);
        bf[ks] = ldB(Vh + (nt * 16) * 128 + ks * 32, 128, lane);
      }
      v8f c = {};
#pragma unroll
      for (int ks = 0; ks < 4; ++ks) c = WMMA_F16(af[ks], bf[ks], c);
      int n0 = mt * 16 + ((lane >> 4) << 3);
      int dd = nt * 16 + (lane & 15);
      if (dd < 24) {
#pragma unroll
        for (int r = 0; r < 8; ++r) AC[(n0 + r) * 96 + h * 24 + dd] = (_Float16)c[r];
      }
    }
    __syncthreads();
  }

  // ---- phase 4: proj GEMM + residual-1 (re-read x at original coord) ----
  for (int t = wave; t < 7 * 6; t += WAVES) {
    int mt = t / 6, nt = t % 6;
    v16h af[3], bf[3];
#pragma unroll
    for (int ks = 0; ks < 3; ++ks) {
      af[ks] = ldA(AC + mt * 16 * 96 + ks * 32, 96, lane);
      bf[ks] = ldB(wproj + (nt * 16) * 96 + ks * 32, 96, lane);
    }
    v8f c = {};
#pragma unroll
    for (int ks = 0; ks < 3; ++ks) c = WMMA_F16(af[ks], bf[ks], c);
    int n0 = mt * 16 + ((lane >> 4) << 3);
    int cc = nt * 16 + (lane & 15);
    float bias = proj_b[cc];
#pragma unroll
    for (int r = 0; r < 8; ++r) {
      int n = n0 + r;
      float resid = 0.0f;
      if (n < 98) resid = x[tok_base(b, wdi, whi, wwi, n) + cc * 50176];
      Y[n * 96 + cc] = resid + c[r] + bias;
    }
  }
  __syncthreads();

  // ---- phase 5: LN2 ----
  if (tid < 112) {
    int n = tid;
    float s1 = 0.f, s2 = 0.f;
    for (int c = 0; c < 96; ++c) { float v = Y[n * 96 + c]; s1 += v; s2 += v * v; }
    float mu = s1 * (1.0f / 96.0f);
    float var = s2 * (1.0f / 96.0f) - mu * mu;
    float rstd = rsqrtf(var + 1e-5f);
    for (int c = 0; c < 96; ++c)
      H2[n * 96 + c] = (_Float16)((Y[n * 96 + c] - mu) * rstd * g2[c] + b2[c]);
  }
  __syncthreads();

  // ---- phase 6: fc1 GEMM + exact GELU ----
  for (int t = wave; t < 7 * 24; t += WAVES) {
    int mt = t / 24, nt = t % 24;
    v16h af[3], bf[3];
#pragma unroll
    for (int ks = 0; ks < 3; ++ks) {
      af[ks] = ldA(H2 + mt * 16 * 96 + ks * 32, 96, lane);
      bf[ks] = ldB(wfc1 + (nt * 16) * 96 + ks * 32, 96, lane);
    }
    v8f c = {};
#pragma unroll
    for (int ks = 0; ks < 3; ++ks) c = WMMA_F16(af[ks], bf[ks], c);
    int n0 = mt * 16 + ((lane >> 4) << 3);
    int j  = nt * 16 + (lane & 15);
    float bias = fc1_b[j];
#pragma unroll
    for (int r = 0; r < 8; ++r) {
      float v = c[r] + bias;
      v = 0.5f * v * (1.0f + erff(v * 0.70710678f));
      F1[(n0 + r) * 384 + j] = (_Float16)v;
    }
  }
  __syncthreads();

  // ---- phase 7: fc2 GEMM + residual-2, scatter to output ----
  for (int t = wave; t < 7 * 6; t += WAVES) {
    int mt = t / 6, nt = t % 6;
    v8f c = {};
#pragma unroll
    for (int kk = 0; kk < 12; kk += 4) {
      v16h af[4], bf[4];
#pragma unroll
      for (int i = 0; i < 4; ++i) {
        af[i] = ldA(F1 + mt * 16 * 384 + (kk + i) * 32, 384, lane);
        bf[i] = ldB(wfc2 + (nt * 16) * 384 + (kk + i) * 32, 384, lane);
      }
#pragma unroll
      for (int i = 0; i < 4; ++i) c = WMMA_F16(af[i], bf[i], c);
    }
    int n0 = mt * 16 + ((lane >> 4) << 3);
    int cc = nt * 16 + (lane & 15);
    float bias = fc2_b[cc];
#pragma unroll
    for (int r = 0; r < 8; ++r) {
      int n = n0 + r;
      if (n < 98) {
        int base = tok_base(b, wdi, whi, wwi, n);
        out[base + cc * 50176] = Y[n * 96 + cc] + c[r] + bias;
      }
    }
  }
}

extern "C" void kernel_launch(void* const* d_in, const int* in_sizes, int n_in,
                              void* d_out, int out_size, void* d_ws, size_t ws_size,
                              hipStream_t stream) {
  const float* x      = (const float*)d_in[0];
  const float* g1     = (const float*)d_in[1];
  const float* b1     = (const float*)d_in[2];
  const float* g2     = (const float*)d_in[3];
  const float* b2     = (const float*)d_in[4];
  const float* qkv_w  = (const float*)d_in[5];
  const float* qkv_b  = (const float*)d_in[6];
  const float* proj_w = (const float*)d_in[7];
  const float* proj_b = (const float*)d_in[8];
  const float* rpb    = (const float*)d_in[9];
  const float* fc1_w  = (const float*)d_in[10];
  const float* fc1_b  = (const float*)d_in[11];
  const float* fc2_w  = (const float*)d_in[12];
  const float* fc2_b  = (const float*)d_in[13];
  float* outp = (float*)d_out;
  _Float16* wt = (_Float16*)d_ws;   // 110592 halves = 216 KB

  swin_prep_weights<<<(110592 + 255) / 256, 256, 0, stream>>>(qkv_w, proj_w, fc1_w, fc2_w, wt);
  swin3d_block<<<2048, 256, 199488, stream>>>(
      x, g1, b1, g2, b2, qkv_b, proj_b, rpb, fc1_b, fc2_b,
      wt, wt + 27648, wt + 36864, wt + 73728, outp);
}
